// CatanGNNEncoder_88046829568557
// MI455X (gfx1250) — matrix operand order repaired and checked by reading, not compiled, not run on hardware
//
#include <hip/hip_runtime.h>
#include <math.h>

typedef __attribute__((ext_vector_type(16))) _Float16 v16h;
typedef __attribute__((ext_vector_type(8)))  float    v8f;

#define B_ 1024
#define HID_ 128
#define NNODE 145            // 19 hex + 54 vertex + 72 edge
#define ASTR 40              // LDS row stride in halves (32 + 8 pad)

// ---------------------------------------------------------------------------
// Topology: replicate np.random.RandomState(0) draws on-device (MT19937 +
// masked-rejection randint), build dst-sorted CSR per relation into ws.
// Slot layout per relation r: rowptr at topo[r*256 .. +72], csr_src at +96.
// ---------------------------------------------------------------------------
__global__ void build_topo(int* __restrict__ topo) {
  if (threadIdx.x != 0 || blockIdx.x != 0) return;
  unsigned st[624]; int mi;
  st[0] = 0u;
  for (int i = 1; i < 624; ++i) st[i] = 1812433253u * (st[i-1] ^ (st[i-1] >> 30)) + (unsigned)i;
  mi = 624;
  auto next = [&]() -> unsigned {
    if (mi >= 624) {
      for (int i = 0; i < 624; ++i) {
        unsigned y = (st[i] & 0x80000000u) | (st[(i+1)%624] & 0x7fffffffu);
        unsigned n = st[(i+397)%624] ^ (y >> 1);
        if (y & 1u) n ^= 0x9908b0dfu;
        st[i] = n;
      }
      mi = 0;
    }
    unsigned y = st[mi++];
    y ^= y >> 11; y ^= (y << 7) & 0x9d2c5680u; y ^= (y << 15) & 0xefc60000u; y ^= y >> 18;
    return y;
  };
  auto randint = [&](int n) -> int {   // [0, n)
    unsigned rng = (unsigned)(n - 1);
    unsigned mask = rng;
    mask |= mask >> 1; mask |= mask >> 2; mask |= mask >> 4; mask |= mask >> 8; mask |= mask >> 16;
    for (;;) { unsigned v = next() & mask; if (v <= rng) return (int)v; }
  };
  int hv[114], v0[72], v1[72];
  for (int i = 0; i < 114; ++i) hv[i] = randint(54);
  for (int i = 0; i <  72; ++i) v0[i] = randint(54);
  for (int i = 0; i <  72; ++i) { int s = randint(53); v1[i] = (v0[i] + 1 + s) % 54; }

  int es_[144], ed_[144];
  for (int rel = 0; rel < 5; ++rel) {
    int ne, nd;
    if (rel == 0)      { ne = 114; nd = 54; for (int i = 0; i < 114; ++i) { es_[i] = i / 6;  ed_[i] = hv[i]; } }
    else if (rel == 1) { ne = 114; nd = 19; for (int i = 0; i < 114; ++i) { es_[i] = hv[i];  ed_[i] = i / 6; } }
    else if (rel == 2) { ne = 144; nd = 72; for (int i = 0; i < 72; ++i) { es_[2*i] = v0[i]; es_[2*i+1] = v1[i]; ed_[2*i] = i; ed_[2*i+1] = i; } }
    else if (rel == 3) { ne = 144; nd = 54; for (int i = 0; i < 72; ++i) { es_[2*i] = i; es_[2*i+1] = i; ed_[2*i] = v0[i]; ed_[2*i+1] = v1[i]; } }
    else               { ne = 144; nd = 54; for (int i = 0; i < 72; ++i) { es_[i] = v0[i]; ed_[i] = v1[i]; es_[72+i] = v1[i]; ed_[72+i] = v0[i]; } }
    int* rp = topo + rel * 256;
    int* cs = topo + rel * 256 + 96;
    int cnt[73];
    for (int d = 0; d <= nd; ++d) cnt[d] = 0;
    for (int e = 0; e < ne; ++e) cnt[ed_[e] + 1]++;
    for (int d = 0; d < nd; ++d) cnt[d+1] += cnt[d];
    for (int d = 0; d <= nd; ++d) rp[d] = cnt[d];
    int pos[73];
    for (int d = 0; d < nd; ++d) pos[d] = cnt[d];
    for (int e = 0; e < ne; ++e) { int d = ed_[e]; cs[pos[d]++] = es_[e]; }
  }
}

// ---------------------------------------------------------------------------
// Was[k,h] = sum_c W[k, h*128+c]*a_src[h,c]  (and Wad with a_dst), per (l,r).
// ---------------------------------------------------------------------------
__global__ void prep_was(const float* __restrict__ gatW, const float* __restrict__ asrc,
                         const float* __restrict__ adst, float* __restrict__ was, float* __restrict__ wad) {
  int idx = blockIdx.x * 256 + threadIdx.x;          // 15*128*4
  if (idx >= 15 * 128 * 4) return;
  int h = idx & 3, k = (idx >> 2) & 127, lr = idx >> 9;
  const float* Wr = gatW + ((size_t)lr * 128 + k) * 512 + h * 128;
  const float* as = asrc + ((size_t)lr * 4 + h) * 128;
  const float* ad = adst + ((size_t)lr * 4 + h) * 128;
  float s = 0.f, d = 0.f;
  for (int c = 0; c < 128; ++c) { float w = Wr[c]; s += w * as[c]; d += w * ad[c]; }
  was[idx] = s; wad[idx] = d;
}

// WcatT[lr][c][h*128+k] = W[lr][k][h*128+c]  (f16, B-operand laid out col-contiguous in K)
__global__ void prep_wcat(const float* __restrict__ gatW, _Float16* __restrict__ wcat) {
  size_t idx = (size_t)blockIdx.x * 256 + threadIdx.x;   // 15*128*512
  if (idx >= (size_t)15 * 128 * 512) return;
  int hk = (int)(idx & 511), c = (int)((idx >> 9) & 127), lr = (int)(idx >> 16);
  int h = hk >> 7, k = hk & 127;
  wcat[idx] = (_Float16)gatW[((size_t)lr * 128 + k) * 512 + h * 128 + c];
}

// ---------------------------------------------------------------------------
// Input projections: x[b,node,c] = feat @ inW + inb (K = 9/7/5, scalar VALU).
// ---------------------------------------------------------------------------
__global__ void in_proj(const float* __restrict__ hexf, const float* __restrict__ vertf,
                        const float* __restrict__ edgef,
                        const float* __restrict__ hW, const float* __restrict__ hb,
                        const float* __restrict__ vW, const float* __restrict__ vb,
                        const float* __restrict__ eW, const float* __restrict__ eb,
                        float* __restrict__ x) {
  size_t idx = (size_t)blockIdx.x * 256 + threadIdx.x;   // B*145*128
  if (idx >= (size_t)B_ * NNODE * 128) return;
  int c = (int)(idx & 127);
  int node = (int)((idx >> 7) % NNODE);
  int b = (int)((idx >> 7) / NNODE);
  const float *f, *W, *bb; int K;
  if (node < 19)      { f = hexf  + ((size_t)b * 19 + node) * 9;        W = hW; bb = hb; K = 9; }
  else if (node < 73) { f = vertf + ((size_t)b * 54 + (node - 19)) * 7; W = vW; bb = vb; K = 7; }
  else                { f = edgef + ((size_t)b * 72 + (node - 73)) * 5; W = eW; bb = eb; K = 5; }
  float a = bb[c];
  for (int k = 0; k < K; ++k) a += f[k] * W[k * 128 + c];
  x[idx] = a;
}

// ---------------------------------------------------------------------------
// es = x_src @ Was, ed = x_dst @ Wad   ([B,N,4] each)
// ---------------------------------------------------------------------------
template<int NS, int ND, int SB, int DB>
__global__ void esed_k(const float* __restrict__ x, const float* __restrict__ was,
                       const float* __restrict__ wad, float* __restrict__ es, float* __restrict__ ed) {
  int idx = blockIdx.x * 256 + threadIdx.x;
  constexpr int TOTS = B_ * NS * 4;
  constexpr int TOT  = TOTS + B_ * ND * 4;
  if (idx >= TOT) return;
  if (idx < TOTS) {
    int i = idx; int h = i & 3; int n = (i >> 2) % NS; int b = (i >> 2) / NS;
    const float* xr = x + ((size_t)b * NNODE + SB + n) * 128;
    float a = 0.f;
    for (int k = 0; k < 128; ++k) a += xr[k] * was[k * 4 + h];
    es[i] = a;
  } else {
    int i = idx - TOTS; int h = i & 3; int n = (i >> 2) % ND; int b = (i >> 2) / ND;
    const float* xr = x + ((size_t)b * NNODE + DB + n) * 128;
    float a = 0.f;
    for (int k = 0; k < 128; ++k) a += xr[k] * wad[k * 4 + h];
    ed[i] = a;
  }
}

// ---------------------------------------------------------------------------
// Per-dst softmax over incoming CSR edges -> alpha[b][pos][h]
// ---------------------------------------------------------------------------
template<int NS, int ND>
__global__ void attn_k(const float* __restrict__ es, const float* __restrict__ ed,
                       const int* __restrict__ rowptr, const int* __restrict__ csrsrc,
                       float* __restrict__ alpha) {
  int b = blockIdx.x;
  for (int t = threadIdx.x; t < ND * 4; t += blockDim.x) {
    int d = t >> 2, h = t & 3;
    int p0 = rowptr[d], p1 = rowptr[d + 1];
    float edv = ed[((size_t)b * ND + d) * 4 + h];
    float m = -3.0e38f;
    for (int p = p0; p < p1; ++p) {
      float v = es[((size_t)b * NS + csrsrc[p]) * 4 + h] + edv;
      v = v > 0.f ? v : 0.2f * v;
      m = v > m ? v : m;
    }
    float ssum = 0.f;
    for (int p = p0; p < p1; ++p) {
      float v = es[((size_t)b * NS + csrsrc[p]) * 4 + h] + edv;
      v = v > 0.f ? v : 0.2f * v;
      ssum += __expf(v - m);
    }
    float inv = 1.f / (ssum + 1e-16f);
    for (int p = p0; p < p1; ++p) {
      float v = es[((size_t)b * NS + csrsrc[p]) * 4 + h] + edv;
      v = v > 0.f ? v : 0.2f * v;
      alpha[((size_t)b * 144 + p) * 4 + h] = __expf(v - m) * inv;
    }
  }
}

// ---------------------------------------------------------------------------
// z[b,d,h,:] = sum_p alpha[p,h] * x[b, src_p, :]  -> zcat f16 [B*ND, 512]
// ---------------------------------------------------------------------------
template<int NS, int ND, int SB>
__global__ void zagg_k(const float* __restrict__ x, const float* __restrict__ alpha,
                       const int* __restrict__ rowptr, const int* __restrict__ csrsrc,
                       _Float16* __restrict__ zcat) {
  int bd = blockIdx.x;
  int b = bd / ND, d = bd - b * ND;
  int c = threadIdx.x;                 // 128
  float a0 = 0.f, a1 = 0.f, a2 = 0.f, a3 = 0.f;
  int p0 = rowptr[d], p1 = rowptr[d + 1];
  for (int p = p0; p < p1; ++p) {
    int s = csrsrc[p];
    float xs = x[((size_t)b * NNODE + SB + s) * 128 + c];
    const float* al = alpha + ((size_t)b * 144 + p) * 4;
    a0 += al[0] * xs; a1 += al[1] * xs; a2 += al[2] * xs; a3 += al[3] * xs;
  }
  _Float16* zr = zcat + ((size_t)b * ND + d) * 512 + c;
  zr[0]   = (_Float16)a0;
  zr[128] = (_Float16)a1;
  zr[256] = (_Float16)a2;
  zr[384] = (_Float16)a3;
}

// ---------------------------------------------------------------------------
// WMMA GEMM: agg[b, DB+d, :] += (zcat[b*ND+d, 0:512] @ Wcat)[:] / 4 + bias
// Block: 64 rows x 128 cols; 8 waves = 4 row-tiles x 2 col-halves; each wave
// owns 4 accumulator tiles (32 VGPRs) -> low pressure, stationary acc regs.
// 16 K-steps of v_wmma_f32_16x16x32_f16.
// ---------------------------------------------------------------------------
template<int ND, int DB>
__global__ __launch_bounds__(256) void gemm_gat(const _Float16* __restrict__ A,
                                                const _Float16* __restrict__ Bw,  // [128 cols][512 k]
                                                const float* __restrict__ bias,  // [128]
                                                float* __restrict__ agg) {
  __shared__ __align__(16) _Float16 As[64 * ASTR];
  __shared__ __align__(16) _Float16 Bs[128 * ASTR];
  const int tid     = threadIdx.x;
  const int wave    = tid >> 5;
  const int lane    = tid & 31;
  const int r       = lane & 15;
  const int sub     = lane >> 4;
  const int rowTile = wave & 3;        // 4 row tiles of 16
  const int colHalf = wave >> 2;       // 2 column halves of 64
  const int row0    = blockIdx.x * 64; // M = B_*ND is a multiple of 64 -> no guards

  v8f acc[4] = {};

  // A loader: 64 rows x 32 halves; thread t copies 8 halves (16B).
  const int arow = tid >> 2, ach = tid & 3;
  // B loader: 128 cols x 32 halves; thread t copies 16 halves (32B).
  const int brow = tid >> 1, bch = tid & 1;

  for (int ks = 0; ks < 16; ++ks) {
    const int k0 = ks * 32;
    {
      const uint4* sa = (const uint4*)(A + (size_t)(row0 + arow) * 512 + k0 + ach * 8);
      uint4 a0 = sa[0];
      *(uint4*)(As + arow * ASTR + ach * 8) = a0;
      const uint4* sb = (const uint4*)(Bw + (size_t)brow * 512 + k0 + bch * 16);
      uint4 b0 = sb[0], b1 = sb[1];
      uint4* db = (uint4*)(Bs + brow * ASTR + bch * 16);
      db[0] = b0; db[1] = b1;
      if (ks < 15) {
        __builtin_prefetch(A + (size_t)(row0 + arow) * 512 + k0 + 32, 0, 1);
      }
    }
    __syncthreads();

    // A fragment: lane<16 holds row (rowTile*16+r), K 0..15; lane>=16: K 16..31
    union { v16h v; uint4 q[2]; } af;
    {
      const int ar = rowTile * 16 + r;
      af.q[0] = *(const uint4*)(As + ar * ASTR + sub * 16);
      af.q[1] = *(const uint4*)(As + ar * ASTR + sub * 16 + 8);
    }
#pragma unroll
    for (int nt = 0; nt < 4; ++nt) {
      union { v16h v; uint4 q[2]; } bf;
      const int bcol = colHalf * 64 + nt * 16 + r;  // lane<16: K 0..15; lane>=16: K 16..31
      const uint4* bp = (const uint4*)(Bs + bcol * ASTR + sub * 16);
      bf.q[0] = bp[0]; bf.q[1] = bp[1];
      acc[nt] = __builtin_amdgcn_wmma_f32_16x16x32_f16(false, af.v, false, bf.v,
                                                       (short)0, acc[nt], false, false);
    }
    __syncthreads();
  }

  // Epilogue: element (m,n) of a tile -> lane = n + (m>=8)*16, vgpr = m%8
  const float invH = 0.25f;
  const int mbase = row0 + rowTile * 16 + sub * 8;
#pragma unroll
  for (int nt = 0; nt < 4; ++nt) {
    const int col = colHalf * 64 + nt * 16 + r;
    const float bi = bias[col];
#pragma unroll
    for (int vr = 0; vr < 8; ++vr) {
      const int grow = mbase + vr;          // = b*ND + d
      const int b = grow / ND;
      const int d = grow - b * ND;
      const size_t idx = ((size_t)b * NNODE + DB + d) * 128 + col;
      agg[idx] += acc[nt][vr] * invH + bi;
    }
  }
}

// ---------------------------------------------------------------------------
// LayerNorm(agg + x) * g + b, ReLU -> x   (per node row of 128)
// ---------------------------------------------------------------------------
__global__ void ln_relu(float* __restrict__ x, const float* __restrict__ agg,
                        const float* __restrict__ lng, const float* __restrict__ lnb, int l) {
  int row = blockIdx.x * 256 + threadIdx.x;   // B*145
  if (row >= B_ * NNODE) return;
  int node = row % NNODE;
  int t = node < 19 ? 0 : (node < 73 ? 1 : 2);
  const float* g  = lng + ((size_t)l * 3 + t) * 128;
  const float* bb = lnb + ((size_t)l * 3 + t) * 128;
  const float* ar = agg + (size_t)row * 128;
  float* xr = x + (size_t)row * 128;
  float mu = 0.f;
  for (int c = 0; c < 128; ++c) mu += ar[c] + xr[c];
  mu *= (1.f / 128.f);
  float var = 0.f;
  for (int c = 0; c < 128; ++c) { float v = ar[c] + xr[c] - mu; var += v * v; }
  var *= (1.f / 128.f);
  float inv = rsqrtf(var + 1e-5f);
  for (int c = 0; c < 128; ++c) {
    float v = (ar[c] + xr[c] - mu) * inv * g[c] + bb[c];
    xr[c] = v > 0.f ? v : 0.f;
  }
}

// ---------------------------------------------------------------------------
// Pool + player gather -> feat[B,398] (stride 400); then 2-layer MLP.
// ---------------------------------------------------------------------------
__global__ void pool_feat(const float* __restrict__ x, const float* __restrict__ ply,
                          const int* __restrict__ cur, float* __restrict__ feat) {
  int idx = blockIdx.x * 256 + threadIdx.x;
  int b = idx >> 9, j = idx & 511;
  if (b >= B_ || j >= 398) return;
  float v;
  if (j < 128)      { float s = 0.f; for (int n = 0; n < 19; ++n) s += x[((size_t)b * NNODE + n) * 128 + j];            v = s * (1.f / 19.f); }
  else if (j < 256) { int c = j - 128; float s = 0.f; for (int n = 0; n < 54; ++n) s += x[((size_t)b * NNODE + 19 + n) * 128 + c]; v = s * (1.f / 54.f); }
  else if (j < 384) { int c = j - 256; float s = 0.f; for (int n = 0; n < 72; ++n) s += x[((size_t)b * NNODE + 73 + n) * 128 + c]; v = s * (1.f / 72.f); }
  else              { v = ply[((size_t)b * 4 + cur[b]) * 14 + (j - 384)]; }
  feat[(size_t)b * 400 + j] = v;
}

__global__ void mlp1_k(const float* __restrict__ feat, const float* __restrict__ W1,
                       const float* __restrict__ b1, float* __restrict__ hbuf) {
  int idx = blockIdx.x * 256 + threadIdx.x;   // B*128
  int b = idx >> 7, j = idx & 127;
  if (b >= B_) return;
  float a = b1[j];
  const float* fr = feat + (size_t)b * 400;
  for (int k = 0; k < 398; ++k) a += fr[k] * W1[(size_t)k * 128 + j];
  hbuf[idx] = a > 0.f ? a : 0.f;
}

__global__ void mlp2_k(const float* __restrict__ hbuf, const float* __restrict__ W2,
                       const float* __restrict__ b2, float* __restrict__ out) {
  int idx = blockIdx.x * 256 + threadIdx.x;   // B*256
  int b = idx >> 8, o = idx & 255;
  if (b >= B_) return;
  float a = b2[o];
  const float* hr = hbuf + (size_t)b * 128;
  for (int k = 0; k < 128; ++k) a += hr[k] * W2[(size_t)k * 256 + o];
  out[idx] = a;
}

// ---------------------------------------------------------------------------
// Per-relation driver
// ---------------------------------------------------------------------------
template<int NS, int ND, int SB, int DB>
static void run_rel(int l, int r, const int* topo, const float* x, const float* was,
                    const float* wad, const _Float16* wcat, const float* gatb,
                    float* esb, float* edb, float* alphab, _Float16* zcat, float* agg,
                    hipStream_t stream) {
  const int lr = l * 5 + r;
  const float*    was_lr = was  + (size_t)lr * 128 * 4;
  const float*    wad_lr = wad  + (size_t)lr * 128 * 4;
  const _Float16* wc_lr  = wcat + (size_t)lr * 128 * 512;
  const float*    b_lr   = gatb + (size_t)lr * 128;
  const int* rowptr = topo + r * 256;
  const int* csr    = topo + r * 256 + 96;
  const int tot = B_ * (NS + ND) * 4;
  esed_k<NS, ND, SB, DB><<<(tot + 255) / 256, 256, 0, stream>>>(x, was_lr, wad_lr, esb, edb);
  attn_k<NS, ND><<<B_, 128, 0, stream>>>(esb, edb, rowptr, csr, alphab);
  zagg_k<NS, ND, SB><<<B_ * ND, 128, 0, stream>>>(x, alphab, rowptr, csr, zcat);
  gemm_gat<ND, DB><<<16 * ND, 256, 0, stream>>>(zcat, wc_lr, b_lr, agg);
}

extern "C" void kernel_launch(void* const* d_in, const int* in_sizes, int n_in,
                              void* d_out, int out_size, void* d_ws, size_t ws_size,
                              hipStream_t stream) {
  (void)in_sizes; (void)n_in; (void)out_size; (void)ws_size;
  const float* hexf  = (const float*)d_in[0];
  const float* vertf = (const float*)d_in[1];
  const float* edgef = (const float*)d_in[2];
  const float* plyf  = (const float*)d_in[3];
  const float* inhW  = (const float*)d_in[4];
  const float* inhb  = (const float*)d_in[5];
  const float* invW  = (const float*)d_in[6];
  const float* invb  = (const float*)d_in[7];
  const float* ineW  = (const float*)d_in[8];
  const float* ineb  = (const float*)d_in[9];
  const float* gatW  = (const float*)d_in[10];
  const float* gasrc = (const float*)d_in[11];
  const float* gadst = (const float*)d_in[12];
  const float* gatb  = (const float*)d_in[13];
  const float* lng   = (const float*)d_in[14];
  const float* lnb   = (const float*)d_in[15];
  const float* W1    = (const float*)d_in[16];
  const float* b1    = (const float*)d_in[17];
  const float* W2    = (const float*)d_in[18];
  const float* b2    = (const float*)d_in[19];
  const int*   curp  = (const int*)d_in[20];
  float* out = (float*)d_out;

  char* ws = (char*)d_ws;
  size_t off = 0;
  auto alloc = [&](size_t bytes) -> char* {
    char* p = ws + off;
    off += (bytes + 255) & ~(size_t)255;
    return p;
  };
  int*      topo   = (int*)alloc(5 * 256 * sizeof(int));
  float*    was    = (float*)alloc((size_t)15 * 128 * 4 * 4);
  float*    wad    = (float*)alloc((size_t)15 * 128 * 4 * 4);
  _Float16* wcat   = (_Float16*)alloc((size_t)15 * 128 * 512 * 2);
  float*    x      = (float*)alloc((size_t)B_ * NNODE * 128 * 4);
  float*    agg    = (float*)alloc((size_t)B_ * NNODE * 128 * 4);
  float*    esb    = (float*)alloc((size_t)B_ * 72 * 4 * 4);
  float*    edb    = (float*)alloc((size_t)B_ * 72 * 4 * 4);
  float*    alphab = (float*)alloc((size_t)B_ * 144 * 4 * 4);
  _Float16* zcat   = (_Float16*)alloc((size_t)B_ * 72 * 512 * 2);
  float*    feat   = (float*)alloc((size_t)B_ * 400 * 4);
  float*    hbuf   = (float*)alloc((size_t)B_ * 128 * 4);

  build_topo<<<1, 1, 0, stream>>>(topo);
  prep_was<<<(15 * 128 * 4 + 255) / 256, 256, 0, stream>>>(gatW, gasrc, gadst, was, wad);
  prep_wcat<<<(15 * 128 * 512 + 255) / 256, 256, 0, stream>>>(gatW, wcat);
  in_proj<<<(B_ * NNODE * 128 + 255) / 256, 256, 0, stream>>>(
      hexf, vertf, edgef, inhW, inhb, invW, invb, ineW, ineb, x);

  for (int l = 0; l < 3; ++l) {
    hipMemsetAsync(agg, 0, (size_t)B_ * NNODE * 128 * 4, stream);
    // r0: hex(19,@0) -> vertex(54,@19)
    run_rel<19, 54,  0, 19>(l, 0, topo, x, was, wad, wcat, gatb, esb, edb, alphab, zcat, agg, stream);
    // r1: vertex(54,@19) -> hex(19,@0)
    run_rel<54, 19, 19,  0>(l, 1, topo, x, was, wad, wcat, gatb, esb, edb, alphab, zcat, agg, stream);
    // r2: vertex(54,@19) -> edge(72,@73)
    run_rel<54, 72, 19, 73>(l, 2, topo, x, was, wad, wcat, gatb, esb, edb, alphab, zcat, agg, stream);
    // r3: edge(72,@73) -> vertex(54,@19)
    run_rel<72, 54, 73, 19>(l, 3, topo, x, was, wad, wcat, gatb, esb, edb, alphab, zcat, agg, stream);
    // r4: vertex -> vertex
    run_rel<54, 54, 19, 19>(l, 4, topo, x, was, wad, wcat, gatb, esb, edb, alphab, zcat, agg, stream);
    ln_relu<<<(B_ * NNODE + 255) / 256, 256, 0, stream>>>(x, agg, lng, lnb, l);
  }

  pool_feat<<<(B_ * 512 + 255) / 256, 256, 0, stream>>>(x, plyf, curp, feat);
  mlp1_k<<<(B_ * 128 + 255) / 256, 256, 0, stream>>>(feat, W1, b1, hbuf);
  mlp2_k<<<(B_ * 256 + 255) / 256, 256, 0, stream>>>(hbuf, W2, b2, out);
}
